// MultiHeadAttention_63024350101553
// MI455X (gfx1250) — matrix operand order repaired and checked by reading, not compiled, not run on hardware
//
#include <hip/hip_runtime.h>
#include <hip/hip_bf16.h>

// ---------------------------------------------------------------------------
// MI455X (gfx1250) multi-head attention forward, f16 WMMA, flash-attention.
// All matmul FLOPs on v_wmma_f32_16x16x32_f16 (wave32, 16x16 tiles).
// Double-buffered LDS pipelines: one barrier per K-step, global-load latency
// hidden behind WMMA compute.
// ---------------------------------------------------------------------------

typedef __attribute__((ext_vector_type(16))) _Float16     v16h;
typedef __attribute__((ext_vector_type(8)))  float        v8f;
typedef __attribute__((ext_vector_type(4)))  unsigned int u32x4;

union Frag16 { v16h v; u32x4 u[2]; };

#define D_MODEL 1024
#define SEQ     2048
#define BATCH   2
#define NH      16
#define DH      64
#define MROWS   4096   // BATCH * SEQ

#define TILE_HALVES (128 * 40)   // one GEMM LDS buffer (padded rows)
#define KV_HALVES   (64 * 72)    // one attention K/V LDS buffer

__device__ __forceinline__ v8f wmma_f16(v16h a, v16h b, v8f c) {
  // D = A(16x32 f16) * B(32x16 f16) + C(16x16 f32)
  return __builtin_amdgcn_wmma_f32_16x16x32_f16(false, a, false, b, (short)0, c,
                                                false, false);
}

__device__ __forceinline__ v8f zero8() {
  v8f r;
#pragma unroll
  for (int i = 0; i < 8; ++i) r[i] = 0.f;
  return r;
}

// ---------------------------------------------------------------------------
// fp32 -> fp16 elementwise convert (for x)
// ---------------------------------------------------------------------------
__global__ __launch_bounds__(256) void cvt_f32_f16_kernel(
    const float* __restrict__ src, _Float16* __restrict__ dst, int n) {
  int i = blockIdx.x * blockDim.x + threadIdx.x;
  if (i < n) dst[i] = (_Float16)src[i];
}

// ---------------------------------------------------------------------------
// Weight transpose + convert: T[n][k] = (f16) W[k][n]  for 4 weights at once
// ---------------------------------------------------------------------------
__global__ __launch_bounds__(256) void wt_cvt_kernel(
    const float* __restrict__ W0, const float* __restrict__ W1,
    const float* __restrict__ W2, const float* __restrict__ W3,
    _Float16* __restrict__ T0, _Float16* __restrict__ T1,
    _Float16* __restrict__ T2, _Float16* __restrict__ T3) {
  __shared__ float tile[32][33];
  int z = blockIdx.z;
  const float* W = (z == 0) ? W0 : ((z == 1) ? W1 : ((z == 2) ? W2 : W3));
  _Float16*    T = (z == 0) ? T0 : ((z == 1) ? T1 : ((z == 2) ? T2 : T3));
  int bn = blockIdx.x * 32;  // output (n) block
  int bk = blockIdx.y * 32;  // input (k) block
  int tx = threadIdx.x, ty = threadIdx.y;
#pragma unroll
  for (int i = ty; i < 32; i += 8)
    tile[i][tx] = W[(size_t)(bk + i) * D_MODEL + bn + tx];
  __syncthreads();
#pragma unroll
  for (int i = ty; i < 32; i += 8)
    T[(size_t)(bn + i) * D_MODEL + bk + tx] = (_Float16)tile[tx][i];
}

// ---------------------------------------------------------------------------
// Shared WMMA GEMM core: C(128x128 f32) += A(128xK f16) * Bt(128xK f16)^T
// A row-major [M][1024], Bt row-major [N][1024] (i.e. W transposed).
// 256 threads = 8 waves in a 2(M) x 4(N) grid; each wave owns 64x32.
// Double-buffered LDS: global loads for tile i+1 staged in regs while WMMAs
// consume tile i; one barrier per K-step.
// ---------------------------------------------------------------------------
__device__ __forceinline__ void gemm_core(const _Float16* __restrict__ A,
                                          const _Float16* __restrict__ Bt,
                                          _Float16* sA, _Float16* sB,  // 2 bufs
                                          int mTile, int nTile,
                                          v8f (&acc)[4][2]) {
  const int tid  = threadIdx.x;
  const int lane = tid & 31;
  const int wave = tid >> 5;
  const int wm   = wave >> 2;      // 0..1
  const int wn   = wave & 3;       // 0..3
  const int lr   = lane & 15;
  const int hl   = lane >> 4;      // half-wave select

#pragma unroll
  for (int i = 0; i < 4; ++i)
#pragma unroll
    for (int j = 0; j < 2; ++j) acc[i][j] = zero8();

  // Staging geometry: thread covers rows r0 and r0+64 at column q0*8.
  const int r0 = tid >> 2, q0 = tid & 3;
  const _Float16* gA0 = &A[(size_t)(mTile + r0) * D_MODEL + q0 * 8];
  const _Float16* gA1 = gA0 + (size_t)64 * D_MODEL;
  const _Float16* gB0 = &Bt[(size_t)(nTile + r0) * D_MODEL + q0 * 8];
  const _Float16* gB1 = gB0 + (size_t)64 * D_MODEL;
  const int lds0 = r0 * 40 + q0 * 8;
  const int lds1 = (r0 + 64) * 40 + q0 * 8;

  u32x4 ra0, ra1, rb0, rb1;

  // Prologue: tile 0 -> buffer 0
  ra0 = *(const u32x4*)(gA0);
  ra1 = *(const u32x4*)(gA1);
  rb0 = *(const u32x4*)(gB0);
  rb1 = *(const u32x4*)(gB1);
  *(u32x4*)&sA[lds0] = ra0;
  *(u32x4*)&sA[lds1] = ra1;
  *(u32x4*)&sB[lds0] = rb0;
  *(u32x4*)&sB[lds1] = rb1;

  int cur = 0;
  const int NIT = D_MODEL / 32;  // 32 K-steps
  for (int it = 0; it < NIT; ++it) {
    // (a) issue global loads for next tile (latency hidden by compute below)
    const bool more = (it + 1 < NIT);
    if (more) {
      const int kn = (it + 1) * 32;
      ra0 = *(const u32x4*)(gA0 + kn);
      ra1 = *(const u32x4*)(gA1 + kn);
      rb0 = *(const u32x4*)(gB0 + kn);
      rb1 = *(const u32x4*)(gB1 + kn);
      if (it + 2 < NIT) {  // gfx1250 global_prefetch_b8 two tiles ahead
        __builtin_prefetch(gA0 + kn + 32, 0, 1);
        __builtin_prefetch(gB0 + kn + 32, 0, 1);
      }
    }

    // (b) single barrier: prev stores visible, prev reads of buf[next] done
    __syncthreads();

    // (c) compute from buffer `cur`
    const _Float16* cA = sA + cur * TILE_HALVES;
    const _Float16* cB = sB + cur * TILE_HALVES;
    Frag16 a[4], b[2];
    const int k0a = hl * 8;
#pragma unroll
    for (int i = 0; i < 4; ++i) {
      // A frag: lane holds row M=lr, k = 16*(e>=8) + 8*hl + (e&7)
      const _Float16* p = &cA[(wm * 64 + i * 16 + lr) * 40 + k0a];
      a[i].u[0] = *(const u32x4*)p;
      a[i].u[1] = *(const u32x4*)(p + 16);
    }
    const int kb = hl * 8;
#pragma unroll
    for (int j = 0; j < 2; ++j) {
      // B frag: lane holds col N=lr, k = 16*(e>=8) + 8*hl + (e&7)
      const _Float16* p = &cB[(wn * 32 + j * 16 + lr) * 40 + kb];
      b[j].u[0] = *(const u32x4*)p;
      b[j].u[1] = *(const u32x4*)(p + 16);
    }
#pragma unroll
    for (int i = 0; i < 4; ++i)
#pragma unroll
      for (int j = 0; j < 2; ++j)
        acc[i][j] = wmma_f16(a[i].v, b[j].v, acc[i][j]);

    // (d) store staged regs into the other buffer
    if (more) {
      _Float16* nA = sA + (cur ^ 1) * TILE_HALVES;
      _Float16* nB = sB + (cur ^ 1) * TILE_HALVES;
      *(u32x4*)&nA[lds0] = ra0;
      *(u32x4*)&nA[lds1] = ra1;
      *(u32x4*)&nB[lds0] = rb0;
      *(u32x4*)&nB[lds1] = rb1;
      cur ^= 1;
    }
  }
}

// ---------------------------------------------------------------------------
// QKV projection: y = x @ W + b, epilogue scatters per-head.
//   z=0: Qh [B,H,S,Dh] f16     z=1: Kh [B,H,S,Dh] f16
//   z=2: Vt [B,H,Dh,S] f16 (transposed so PV B-frags are contiguous)
// ---------------------------------------------------------------------------
__global__ __launch_bounds__(256) void gemm_qkv_kernel(
    const _Float16* __restrict__ A,
    const _Float16* __restrict__ Btq, const _Float16* __restrict__ Btk,
    const _Float16* __restrict__ Btv,
    const float* __restrict__ bq, const float* __restrict__ bk,
    const float* __restrict__ bv,
    _Float16* __restrict__ Qh, _Float16* __restrict__ Kh,
    _Float16* __restrict__ Vth) {
  __shared__ _Float16 sA[2 * TILE_HALVES];
  __shared__ _Float16 sB[2 * TILE_HALVES];
  const int z = blockIdx.z;
  const _Float16* Bt  = (z == 0) ? Btq : ((z == 1) ? Btk : Btv);
  const float*    bia = (z == 0) ? bq  : ((z == 1) ? bk  : bv);
  const int nTile = blockIdx.x * 128;
  const int mTile = blockIdx.y * 128;

  v8f acc[4][2];
  gemm_core(A, Bt, sA, sB, mTile, nTile, acc);

  const int tid = threadIdx.x, lane = tid & 31, wave = tid >> 5;
  const int wm = wave >> 2, wn = wave & 3, lr = lane & 15, hl = lane >> 4;
#pragma unroll
  for (int i = 0; i < 4; ++i) {
    int mBase = mTile + wm * 64 + i * 16 + hl * 8;  // C rows: v + 8*hl
#pragma unroll
    for (int j = 0; j < 2; ++j) {
      int n = nTile + wn * 32 + j * 16 + lr;        // C cols: lr
      float bn = bia[n];
      int h = n >> 6, d = n & 63;
#pragma unroll
      for (int v = 0; v < 8; ++v) {
        int mm = mBase + v;
        int b_ = mm >> 11, s = mm & 2047;
        float val = acc[i][j][v] + bn;
        if (z == 0)
          Qh[(size_t)((b_ * NH + h) * SEQ + s) * DH + d] = (_Float16)val;
        else if (z == 1)
          Kh[(size_t)((b_ * NH + h) * SEQ + s) * DH + d] = (_Float16)val;
        else
          Vth[(size_t)((b_ * NH + h) * DH + d) * SEQ + s] = (_Float16)val;
      }
    }
  }
}

// ---------------------------------------------------------------------------
// Output projection: out = ctx @ Wo + bo  (f32 result into d_out)
// ---------------------------------------------------------------------------
__global__ __launch_bounds__(256) void gemm_out_kernel(
    const _Float16* __restrict__ A, const _Float16* __restrict__ Bt,
    const float* __restrict__ bia, float* __restrict__ Out) {
  __shared__ _Float16 sA[2 * TILE_HALVES];
  __shared__ _Float16 sB[2 * TILE_HALVES];
  const int nTile = blockIdx.x * 128;
  const int mTile = blockIdx.y * 128;

  v8f acc[4][2];
  gemm_core(A, Bt, sA, sB, mTile, nTile, acc);

  const int tid = threadIdx.x, lane = tid & 31, wave = tid >> 5;
  const int wm = wave >> 2, wn = wave & 3, lr = lane & 15, hl = lane >> 4;
#pragma unroll
  for (int i = 0; i < 4; ++i) {
    int mBase = mTile + wm * 64 + i * 16 + hl * 8;
#pragma unroll
    for (int j = 0; j < 2; ++j) {
      int n = nTile + wn * 32 + j * 16 + lr;
      float bn = bia[n];
#pragma unroll
      for (int v = 0; v < 8; ++v)
        Out[(size_t)(mBase + v) * D_MODEL + n] = acc[i][j][v] + bn;
    }
  }
}

// ---------------------------------------------------------------------------
// Flash attention: one block = 64 queries of one (b,h); 4 waves x 16 queries.
// Online softmax over 64-key blocks; scores and PV both on WMMA.
// Double-buffered K/V staging, one barrier per key-block.
// ---------------------------------------------------------------------------
__global__ __launch_bounds__(128) void attn_kernel(
    const _Float16* __restrict__ Qh, const _Float16* __restrict__ Kh,
    const _Float16* __restrict__ Vth, _Float16* __restrict__ ctx) {
  __shared__ _Float16 sK[2 * KV_HALVES];   // K tile [key][dh]
  __shared__ _Float16 sV[2 * KV_HALVES];   // Vt tile [dh][key]
  __shared__ _Float16 sP[4 * 16 * 72];     // per-wave prob scratch [16][72]

  const int bh    = blockIdx.y;              // b*NH + h
  const int qBase = blockIdx.x * 64;
  const int tid  = threadIdx.x;
  const int lane = tid & 31;
  const int wave = tid >> 5;                 // 0..3
  const int lr   = lane & 15;
  const int hl   = lane >> 4;

  const _Float16* Q = Qh  + (size_t)bh * SEQ * DH;
  const _Float16* K = Kh  + (size_t)bh * SEQ * DH;
  const _Float16* V = Vth + (size_t)bh * DH * SEQ;

  const float SCL = 0.125f * 1.44269504088896f;  // log2(e)/sqrt(Dh)

  // This wave's Q fragments (16 queries x 64 dh = 2 A-frags) kept in regs
  Frag16 qa[2];
  {
    const int qrow = qBase + wave * 16 + lr;
    const int k0a  = hl * 8;
#pragma unroll
    for (int ks = 0; ks < 2; ++ks) {
      const _Float16* p = &Q[(size_t)qrow * DH + ks * 32 + k0a];
      qa[ks].u[0] = *(const u32x4*)p;
      qa[ks].u[1] = *(const u32x4*)(p + 16);
    }
  }

  // Staging geometry: thread covers rows rk+16t (t=0..3) at column qk*8.
  const int rk = tid >> 3, qk = tid & 7;
  u32x4 kreg[4], vreg[4];

  float mOld[8], lSum[8];
  v8f o[4];
#pragma unroll
  for (int v = 0; v < 8; ++v) { mOld[v] = -1.0e30f; lSum[v] = 0.f; }
#pragma unroll
  for (int j = 0; j < 4; ++j) o[j] = zero8();

  // Prologue: key-block 0 -> buffer 0
#pragma unroll
  for (int t = 0; t < 4; ++t) {
    int r = rk + 16 * t;
    kreg[t] = *(const u32x4*)&K[(size_t)r * DH + qk * 8];
    vreg[t] = *(const u32x4*)&V[(size_t)r * SEQ + qk * 8];
  }
#pragma unroll
  for (int t = 0; t < 4; ++t) {
    int r = rk + 16 * t;
    *(u32x4*)&sK[r * 72 + qk * 8] = kreg[t];
    *(u32x4*)&sV[r * 72 + qk * 8] = vreg[t];
  }

  int cur = 0;
  const int NKB = SEQ / 64;  // 32 key blocks
  for (int it = 0; it < NKB; ++it) {
    // (a) issue global loads for next key block
    const bool more = (it + 1 < NKB);
    if (more) {
      const int kb = (it + 1) * 64;
#pragma unroll
      for (int t = 0; t < 4; ++t) {
        int r = rk + 16 * t;
        kreg[t] = *(const u32x4*)&K[(size_t)(kb + r) * DH + qk * 8];
        vreg[t] = *(const u32x4*)&V[(size_t)r * SEQ + kb + qk * 8];
      }
      if (it + 2 < NKB)
        __builtin_prefetch(&K[(size_t)(kb + 64 + rk) * DH + qk * 8], 0, 1);
    }

    // (b) single barrier per key block
    __syncthreads();

    const _Float16* cK = sK + cur * KV_HALVES;
    const _Float16* cV = sV + cur * KV_HALVES;

    // ---- scores: S = Q (16x64) . K^T  -> 16 x 64, 8 WMMAs ----
    v8f c[4];
#pragma unroll
    for (int j = 0; j < 4; ++j) c[j] = zero8();
#pragma unroll
    for (int j = 0; j < 4; ++j) {
      int n = j * 16 + lr;  // key within tile
#pragma unroll
      for (int ks = 0; ks < 2; ++ks) {
        Frag16 bf;
        const _Float16* p = &cK[n * 72 + ks * 32 + hl * 8];
        bf.u[0] = *(const u32x4*)p;
        bf.u[1] = *(const u32x4*)(p + 16);
        c[j] = wmma_f16(qa[ks].v, bf.v, c[j]);
      }
    }

    // ---- online softmax (rows = v + 8*hl, cols across 16 lanes) ----
    float mNew[8], alpha[8], ps[8];
#pragma unroll
    for (int v = 0; v < 8; ++v) {
      float mx = fmaxf(fmaxf(c[0][v], c[1][v]), fmaxf(c[2][v], c[3][v]));
#pragma unroll
      for (int d = 1; d < 16; d <<= 1) mx = fmaxf(mx, __shfl_xor(mx, d, 32));
      mNew[v]  = fmaxf(mOld[v], mx);
      alpha[v] = exp2f((mOld[v] - mNew[v]) * SCL);
      ps[v]    = 0.f;
    }
    _Float16* pw = &sP[wave * 16 * 72];
#pragma unroll
    for (int j = 0; j < 4; ++j)
#pragma unroll
      for (int v = 0; v < 8; ++v) {
        float p = exp2f((c[j][v] - mNew[v]) * SCL);
        ps[v] += p;
        pw[(v + hl * 8) * 72 + j * 16 + lr] = (_Float16)p;  // C-layout -> LDS
      }
#pragma unroll
    for (int v = 0; v < 8; ++v) {
#pragma unroll
      for (int d = 1; d < 16; d <<= 1) ps[v] += __shfl_xor(ps[v], d, 32);
      lSum[v] = lSum[v] * alpha[v] + ps[v];
      mOld[v] = mNew[v];
    }
#pragma unroll
    for (int j = 0; j < 4; ++j)
#pragma unroll
      for (int v = 0; v < 8; ++v) o[j][v] *= alpha[v];

    // ---- O += P (16x64) . V  -> 16 x 64, 8 WMMAs ----
    // LDS ops are in-order within a wave, so the P stores above are visible.
    Frag16 af[2];
#pragma unroll
    for (int ks = 0; ks < 2; ++ks) {
      const _Float16* p = &pw[lr * 72 + ks * 32 + hl * 8];  // A-layout read
      af[ks].u[0] = *(const u32x4*)p;
      af[ks].u[1] = *(const u32x4*)(p + 16);
    }
#pragma unroll
    for (int j = 0; j < 4; ++j) {
      int n = j * 16 + lr;  // dh
#pragma unroll
      for (int ks = 0; ks < 2; ++ks) {
        Frag16 bf;
        const _Float16* p = &cV[n * 72 + ks * 32 + hl * 8];
        bf.u[0] = *(const u32x4*)p;
        bf.u[1] = *(const u32x4*)(p + 16);
        o[j] = wmma_f16(af[ks].v, bf.v, o[j]);
      }
    }

    // (d) store staged regs into the other buffer
    if (more) {
      _Float16* nK = sK + (cur ^ 1) * KV_HALVES;
      _Float16* nV = sV + (cur ^ 1) * KV_HALVES;
#pragma unroll
      for (int t = 0; t < 4; ++t) {
        int r = rk + 16 * t;
        *(u32x4*)&nK[r * 72 + qk * 8] = kreg[t];
        *(u32x4*)&nV[r * 72 + qk * 8] = vreg[t];
      }
      cur ^= 1;
    }
  }

  // ---- epilogue: normalize and write ctx as [B,S,D] f16 ----
  const int b_ = bh >> 4, h = bh & 15;
#pragma unroll
  for (int j = 0; j < 4; ++j) {
    int d = j * 16 + lr;
#pragma unroll
    for (int v = 0; v < 8; ++v) {
      int s = qBase + wave * 16 + v + hl * 8;
      float val = o[j][v] / lSum[v];
      ctx[(size_t)(b_ * SEQ + s) * D_MODEL + h * DH + d] = (_Float16)val;
    }
  }
}

// ---------------------------------------------------------------------------
// Host launch
// ---------------------------------------------------------------------------
extern "C" void kernel_launch(void* const* d_in, const int* in_sizes, int n_in,
                              void* d_out, int out_size, void* d_ws,
                              size_t ws_size, hipStream_t stream) {
  (void)in_sizes; (void)n_in; (void)out_size; (void)ws_size;
  const float* x  = (const float*)d_in[0];
  const float* Wq = (const float*)d_in[1];
  const float* bq = (const float*)d_in[2];
  const float* Wk = (const float*)d_in[3];
  const float* bk = (const float*)d_in[4];
  const float* Wv = (const float*)d_in[5];
  const float* bv = (const float*)d_in[6];
  const float* Wo = (const float*)d_in[7];
  const float* bo = (const float*)d_in[8];

  char* ws = (char*)d_ws;
  // Workspace layout (48 MB total)
  _Float16* xh   = (_Float16*)(ws + 0);                 // 8 MB  [4096][1024]
  _Float16* Wtq  = (_Float16*)(ws + (8u  << 20));       // 2 MB  [1024][1024]
  _Float16* Wtk  = (_Float16*)(ws + (10u << 20));
  _Float16* Wtv  = (_Float16*)(ws + (12u << 20));
  _Float16* Wto  = (_Float16*)(ws + (14u << 20));
  _Float16* Qh   = (_Float16*)(ws + (16u << 20));       // 8 MB  [B,H,S,Dh]
  _Float16* Kh   = (_Float16*)(ws + (24u << 20));       // 8 MB
  _Float16* Vth  = (_Float16*)(ws + (32u << 20));       // 8 MB  [B,H,Dh,S]
  _Float16* ctxh = (_Float16*)(ws + (40u << 20));       // 8 MB  [B,S,D]

  // 1) convert x to f16
  cvt_f32_f16_kernel<<<(MROWS * D_MODEL) / 256, 256, 0, stream>>>(
      x, xh, MROWS * D_MODEL);

  // 2) transpose+convert the four weights
  wt_cvt_kernel<<<dim3(32, 32, 4), dim3(32, 8), 0, stream>>>(
      Wq, Wk, Wv, Wo, Wtq, Wtk, Wtv, Wto);

  // 3) QKV projections (z selects q/k/v)
  gemm_qkv_kernel<<<dim3(D_MODEL / 128, MROWS / 128, 3), 256, 0, stream>>>(
      xh, Wtq, Wtk, Wtv, bq, bk, bv, Qh, Kh, Vth);

  // 4) flash attention
  attn_kernel<<<dim3(SEQ / 64, BATCH * NH), 128, 0, stream>>>(Qh, Kh, Vth, ctxh);

  // 5) output projection into d_out (f32)
  gemm_out_kernel<<<dim3(D_MODEL / 128, MROWS / 128), 256, 0, stream>>>(
      ctxh, Wto, bo, (float*)d_out);
}
